// SpatialAttention_68513318306527
// MI455X (gfx1250) — matrix operand order repaired and checked by reading, not compiled, not run on hardware
//
#include <hip/hip_runtime.h>

// Problem constants (match reference)
#define B_    2
#define T_    2048
#define C_    1024
#define H_    16
#define DH_   64
#define N3C   3072
#define BH    (B_*H_)      // 32
#define ROWS  (B_*T_)      // 4096
#define NROWS (BH*T_)      // 65536 softmax rows

typedef __attribute__((ext_vector_type(16))) __bf16 v16bf;
typedef __attribute__((ext_vector_type(8)))  __bf16 v8bf;
typedef __attribute__((ext_vector_type(8)))  float  v8f;
typedef __attribute__((ext_vector_type(2)))  float  f2;

// ---------------- WMMA fragment helpers (per CDNA5 ISA layouts) ---

// A-matrix 16x32 bf16, row-major source, ld elements.
// lane: M = lane&15 ; K-base = (lane>>4)*8 ; elems 0..7 -> K+0..7, 8..15 -> K+16..23
__device__ __forceinline__ v16bf load_a_bf16(const __bf16* __restrict__ base,
                                             int ld, int m0, int k0, int lane) {
  int m  = m0 + (lane & 15);
  int kb = k0 + ((lane >> 4) << 3);
  const __bf16* p = base + (size_t)m * ld + kb;
  v8bf lo = *reinterpret_cast<const v8bf*>(p);
  v8bf hi = *reinterpret_cast<const v8bf*>(p + 16);
  v16bf a;
#pragma unroll
  for (int i = 0; i < 8; i++) { a[i] = lo[i]; a[8 + i] = hi[i]; }
  return a;
}

// A-matrix 16x32 from fp32 source (attn), converted to bf16 in regs.
// Source base may be only 8-byte aligned -> use float2 loads.
__device__ __forceinline__ v16bf load_a_f32(const float* __restrict__ base,
                                            int ld, int m0, int k0, int lane) {
  int m  = m0 + (lane & 15);
  int kb = k0 + ((lane >> 4) << 3);
  const float* p = base + (size_t)m * ld + kb;
  v16bf a;
#pragma unroll
  for (int j = 0; j < 4; j++) {
    f2 lo = *reinterpret_cast<const f2*>(p + 2 * j);
    f2 hi = *reinterpret_cast<const f2*>(p + 16 + 2 * j);
    a[2 * j]     = (__bf16)lo[0];
    a[2 * j + 1] = (__bf16)lo[1];
    a[8 + 2 * j]     = (__bf16)hi[0];
    a[8 + 2 * j + 1] = (__bf16)hi[1];
  }
  return a;
}

// B-matrix 32x16 bf16 from TRANSPOSED source (N x K row-major), ld elements.
// lane: N = lane&15 ; K-base = (lane>>4)*8 ; elems 0..7 -> K+0..7, 8..15 -> K+16..23
__device__ __forceinline__ v16bf load_b_bf16T(const __bf16* __restrict__ baseT,
                                              int ld, int n0, int k0, int lane) {
  int n  = n0 + (lane & 15);
  int kb = k0 + ((lane >> 4) << 3);
  const __bf16* p = baseT + (size_t)n * ld + kb;
  v8bf lo = *reinterpret_cast<const v8bf*>(p);
  v8bf hi = *reinterpret_cast<const v8bf*>(p + 16);
  v16bf b;
#pragma unroll
  for (int i = 0; i < 8; i++) { b[i] = lo[i]; b[8 + i] = hi[i]; }
  return b;
}

__device__ __forceinline__ v8f wmma_bf16(v16bf a, v16bf b, v8f c) {
  return __builtin_amdgcn_wmma_f32_16x16x32_bf16(false, a, false, b,
                                                 (short)0, c, false, false);
}

// ---------------- Prep kernels ----------------------------------------------

__global__ void cvt_bf16_kernel(const float* __restrict__ src,
                                __bf16* __restrict__ dst, int n) {
  int i = blockIdx.x * 256 + threadIdx.x;
  if (i < n) dst[i] = (__bf16)src[i];
}

// dstT[n*K + k] = src[k*N + n]
__global__ void cvt_transpose_kernel(const float* __restrict__ src,
                                     __bf16* __restrict__ dstT, int K, int N) {
  int i = blockIdx.x * 256 + threadIdx.x;
  if (i < K * N) {
    int k = i / N, n = i % N;
    dstT[(size_t)n * K + k] = (__bf16)src[i];
  }
}

__global__ void copy_f32_kernel(const float* __restrict__ src,
                                float* __restrict__ dst, int n) {
  int i = blockIdx.x * 256 + threadIdx.x;
  if (i < n) dst[i] = src[i];
}

// ---------------- QKV projection GEMM: (4096x1024) @ (1024x3072) ------------
// Writes q,k row-major bf16 (BH,T,Dh) and v transposed bf16 (BH,Dh,T).
__global__ __launch_bounds__(256) void qkv_gemm_kernel(
    const __bf16* __restrict__ xb, const __bf16* __restrict__ wT,
    __bf16* __restrict__ qb, __bf16* __restrict__ kb, __bf16* __restrict__ vT) {
  int wave = (blockIdx.x * 256 + threadIdx.x) >> 5;
  int lane = threadIdx.x & 31;
  int tn = wave % (N3C / 16);      // 192 tiles across N
  int tm = wave / (N3C / 16);      // 256 tiles across M
  int m0 = tm * 16, n0 = tn * 16;
  v8f acc = {};
  for (int k0 = 0; k0 < C_; k0 += 32) {
    v16bf a = load_a_bf16(xb, C_, m0, k0, lane);
    v16bf b = load_b_bf16T(wT, C_, n0, k0, lane);
    acc = wmma_bf16(a, b, acc);
  }
  int col = n0 + (lane & 15);
  int sec = col >> 10;             // 0=q 1=k 2=v (uniform per tile)
  int cc  = col & (C_ - 1);
  int h = cc >> 6, d = cc & 63;
  int rbase = m0 + ((lane >> 4) << 3);
#pragma unroll
  for (int r = 0; r < 8; r++) {
    int row = rbase + r;
    int b_ = row >> 11, t = row & (T_ - 1);
    int bh = b_ * H_ + h;
    __bf16 val = (__bf16)acc[r];
    if (sec == 0)      qb[(((size_t)bh * T_ + t) << 6) + d] = val;
    else if (sec == 1) kb[(((size_t)bh * T_ + t) << 6) + d] = val;
    else               vT[((size_t)bh * DH_ + d) * T_ + t] = val;
  }
}

// ---------------- Logits: q @ k^T / 8 - dists (lower-triangle tiles only) ---
__global__ __launch_bounds__(256) void logits_kernel(
    const __bf16* __restrict__ qb, const __bf16* __restrict__ kb,
    const float* __restrict__ dists, float* __restrict__ attn) {
  int wave = (blockIdx.x * 256 + threadIdx.x) >> 5;
  int lane = threadIdx.x & 31;
  int ki = wave & 127;
  int qi = (wave >> 7) & 127;
  int bh = wave >> 14;
  if (ki > qi) return;             // fully-masked tile: softmax never reads it
  const __bf16* Q = qb + (size_t)bh * T_ * DH_;
  const __bf16* K = kb + (size_t)bh * T_ * DH_;
  int m0 = qi * 16, n0 = ki * 16;
  v8f acc = {};
#pragma unroll
  for (int k0 = 0; k0 < DH_; k0 += 32) {
    v16bf a = load_a_bf16(Q, DH_, m0, k0, lane);
    v16bf b = load_b_bf16T(K, DH_, n0, k0, lane);
    acc = wmma_bf16(a, b, acc);
  }
  int b_ = bh >> 4;
  const float* D = dists + (size_t)b_ * T_ * T_;
  float* A = attn + (size_t)bh * T_ * T_;
  int n = n0 + (lane & 15);
  int rbase = m0 + ((lane >> 4) << 3);
#pragma unroll
  for (int r = 0; r < 8; r++) {
    int q = rbase + r;
    A[(size_t)q * T_ + n] = acc[r] * 0.125f - D[(size_t)q * T_ + n];
  }
}

// ---------------- Softmax row + per-row energy partials ---------------------
__global__ __launch_bounds__(256) void softmax_kernel(
    float* __restrict__ attn, const float* __restrict__ dists,
    float* __restrict__ dpart, float* __restrict__ epart) {
  int row = blockIdx.x;            // bh*T + q
  int q  = row & (T_ - 1);
  int bh = row >> 11;
  int b_ = bh >> 4;
  float* L = attn + (size_t)row * T_;
  const float* D = dists + ((size_t)b_ * T_ + q) * T_;
  int tid = threadIdx.x;
  __shared__ float sbuf[256];

  float lmax = -3.0e38f;
#pragma unroll
  for (int j = 0; j < 8; j++) {
    int k = j * 256 + tid;
    if (k <= q) lmax = fmaxf(lmax, L[k]);
  }
  sbuf[tid] = lmax; __syncthreads();
  for (int s = 128; s > 0; s >>= 1) {
    if (tid < s) sbuf[tid] = fmaxf(sbuf[tid], sbuf[tid + s]);
    __syncthreads();
  }
  float m = sbuf[0]; __syncthreads();

  float ev[8];
  float lsum = 0.f;
#pragma unroll
  for (int j = 0; j < 8; j++) {
    int k = j * 256 + tid;
    if (k <= q) { float e = __expf(L[k] - m); ev[j] = e; lsum += e; }
    else        ev[j] = 0.f;
  }
  sbuf[tid] = lsum; __syncthreads();
  for (int s = 128; s > 0; s >>= 1) {
    if (tid < s) sbuf[tid] += sbuf[tid + s];
    __syncthreads();
  }
  float inv = 1.0f / sbuf[0]; __syncthreads();

  float dloc = 0.f, eloc = 0.f;
#pragma unroll
  for (int j = 0; j < 8; j++) {
    int k = j * 256 + tid;
    float a = ev[j] * inv;
    L[k] = a;                       // exact 0 above diagonal
    dloc += a * D[k];
    eloc -= a * __logf(a + 1e-9f);  // a==0 contributes 0
  }
  sbuf[tid] = dloc; __syncthreads();
  for (int s = 128; s > 0; s >>= 1) {
    if (tid < s) sbuf[tid] += sbuf[tid + s];
    __syncthreads();
  }
  if (tid == 0) dpart[row] = sbuf[0];
  __syncthreads();
  sbuf[tid] = eloc; __syncthreads();
  for (int s = 128; s > 0; s >>= 1) {
    if (tid < s) sbuf[tid] += sbuf[tid + s];
    __syncthreads();
  }
  if (tid == 0) epart[row] = sbuf[0];
}

// ---------------- Deterministic energy reduction ----------------------------
__global__ __launch_bounds__(256) void energy_kernel(
    const float* __restrict__ dpart, const float* __restrict__ epart,
    float* __restrict__ out_d, float* __restrict__ out_e) {
  __shared__ float sd[256], se[256];
  int tid = threadIdx.x;
  float a = 0.f, b = 0.f;
  for (int i = tid; i < NROWS; i += 256) { a += dpart[i]; b += epart[i]; }
  sd[tid] = a; se[tid] = b; __syncthreads();
  for (int s = 128; s > 0; s >>= 1) {
    if (tid < s) { sd[tid] += sd[tid + s]; se[tid] += se[tid + s]; }
    __syncthreads();
  }
  if (tid == 0) { *out_d = sd[0] * (1.0f / NROWS); *out_e = se[0] * (1.0f / NROWS); }
}

// ---------------- attn @ v (K-loop truncated at causal boundary) ------------
__global__ __launch_bounds__(256) void attnv_kernel(
    const float* __restrict__ attn, const __bf16* __restrict__ vT,
    __bf16* __restrict__ aout) {
  int wave = (blockIdx.x * 256 + threadIdx.x) >> 5;
  int lane = threadIdx.x & 31;
  int ni = wave & 3;               // 4 N-tiles (Dh=64)
  int qi = (wave >> 2) & 127;      // 128 M-tiles
  int bh = wave >> 9;              // 32 heads*batch
  const float*  A = attn + (size_t)bh * T_ * T_;
  const __bf16* V = vT   + (size_t)bh * DH_ * T_;
  int m0 = qi * 16, n0 = ni * 16;
  v8f acc = {};
  int kmax = m0 + 15;              // attn is exactly 0 for k > q
  for (int k0 = 0; k0 <= kmax; k0 += 32) {
    v16bf a = load_a_f32(A, T_, m0, k0, lane);
    v16bf b = load_b_bf16T(V, T_, n0, k0, lane);
    acc = wmma_bf16(a, b, acc);
  }
  int b_ = bh >> 4, h = bh & 15;
  int d = n0 + (lane & 15);
  int rbase = m0 + ((lane >> 4) << 3);
#pragma unroll
  for (int r = 0; r < 8; r++) {
    int q = rbase + r;
    aout[((size_t)(b_ * T_ + q)) * C_ + h * DH_ + d] = (__bf16)acc[r];
  }
}

// ---------------- Final projection GEMM: (4096x1024) @ (1024x1024) ----------
__global__ __launch_bounds__(256) void proj_gemm_kernel(
    const __bf16* __restrict__ aout, const __bf16* __restrict__ wT,
    float* __restrict__ out) {
  int wave = (blockIdx.x * 256 + threadIdx.x) >> 5;
  int lane = threadIdx.x & 31;
  int tn = wave & 63;              // 64 N-tiles
  int tm = wave >> 6;              // 256 M-tiles
  int m0 = tm * 16, n0 = tn * 16;
  v8f acc = {};
  for (int k0 = 0; k0 < C_; k0 += 32) {
    v16bf a = load_a_bf16(aout, C_, m0, k0, lane);
    v16bf b = load_b_bf16T(wT, C_, n0, k0, lane);
    acc = wmma_bf16(a, b, acc);
  }
  int n = n0 + (lane & 15);
  int rbase = m0 + ((lane >> 4) << 3);
#pragma unroll
  for (int r = 0; r < 8; r++) {
    int row = rbase + r;
    out[(size_t)row * C_ + n] = acc[r];
  }
}

// ---------------- Host launcher ---------------------------------------------
extern "C" void kernel_launch(void* const* d_in, const int* in_sizes, int n_in,
                              void* d_out, int out_size, void* d_ws, size_t ws_size,
                              hipStream_t stream) {
  (void)in_sizes; (void)n_in; (void)out_size; (void)ws_size;
  const float* x     = (const float*)d_in[0];
  const float* dists = (const float*)d_in[1];
  const float* Wqkv  = (const float*)d_in[2];
  const float* Wproj = (const float*)d_in[3];
  float* out = (float*)d_out;

  // Output layout (flat fp32, reference return order)
  const size_t off_proj  = 0;                               // B*T*C
  const size_t off_de    = (size_t)ROWS * C_;               // 4,194,304
  const size_t off_fe    = off_de + 1;
  const size_t off_attn  = off_fe + 1;                      // B*H*T*T
  const size_t off_dists = off_attn + (size_t)BH * T_ * T_; // B*T*T

  // Workspace layout
  char* ws = (char*)d_ws;
  __bf16* xb     = (__bf16*)ws; ws += (size_t)ROWS * C_  * 2;  // 8 MB
  __bf16* wqkvT  = (__bf16*)ws; ws += (size_t)C_   * N3C * 2;  // 6 MB
  __bf16* wprojT = (__bf16*)ws; ws += (size_t)C_   * C_  * 2;  // 2 MB
  __bf16* qb     = (__bf16*)ws; ws += (size_t)BH * T_ * DH_ * 2; // 8 MB
  __bf16* kb     = (__bf16*)ws; ws += (size_t)BH * T_ * DH_ * 2; // 8 MB
  __bf16* vT     = (__bf16*)ws; ws += (size_t)BH * DH_ * T_ * 2; // 8 MB
  __bf16* aout   = (__bf16*)ws; ws += (size_t)ROWS * C_ * 2;     // 8 MB
  float*  dpart  = (float*)ws;  ws += (size_t)NROWS * 4;         // 256 KB
  float*  epart  = (float*)ws;  ws += (size_t)NROWS * 4;         // 256 KB

  // 1) bf16 conversions + transposes + dists copy
  cvt_bf16_kernel<<<(ROWS * C_) / 256, 256, 0, stream>>>(x, xb, ROWS * C_);
  cvt_transpose_kernel<<<(C_ * N3C) / 256, 256, 0, stream>>>(Wqkv, wqkvT, C_, N3C);
  cvt_transpose_kernel<<<(C_ * C_) / 256, 256, 0, stream>>>(Wproj, wprojT, C_, C_);
  copy_f32_kernel<<<(B_ * T_ * T_) / 256, 256, 0, stream>>>(dists, out + off_dists,
                                                            B_ * T_ * T_);
  // 2) QKV projection
  qkv_gemm_kernel<<<(256 * 192) / 8, 256, 0, stream>>>(xb, wqkvT, qb, kb, vT);
  // 3) Masked logits (written into attn region of d_out)
  logits_kernel<<<(BH * 128 * 128) / 8, 256, 0, stream>>>(qb, kb, dists,
                                                          out + off_attn);
  // 4) Softmax + per-row energy partials
  softmax_kernel<<<NROWS, 256, 0, stream>>>(out + off_attn, dists, dpart, epart);
  // 5) Deterministic energy reduction
  energy_kernel<<<1, 256, 0, stream>>>(dpart, epart, out + off_de, out + off_fe);
  // 6) attn @ v
  attnv_kernel<<<(BH * 128 * 4) / 8, 256, 0, stream>>>(out + off_attn, vT, aout);
  // 7) Output projection
  proj_gemm_kernel<<<(256 * 64) / 8, 256, 0, stream>>>(aout, wprojT, out + off_proj);
}
